// TransformerBlock_53111565582806
// MI455X (gfx1250) — matrix operand order repaired and checked by reading, not compiled, not run on hardware
//
#include <hip/hip_runtime.h>

// ---------------- model dims ----------------
#define DD 512
#define HH 8
#define FF 2048
#define LL 6
#define VV 128
#define BBATCH 8
#define SSQ 512
#define DH 64
#define MM (BBATCH * SSQ)   // 4096 token rows

typedef __attribute__((ext_vector_type(16))) __bf16 v16bf;
typedef __attribute__((ext_vector_type(8)))  float  v8f;

__device__ __forceinline__ unsigned short f2bf(float f) {
  unsigned u = __float_as_uint(f);
  u += 0x7FFFu + ((u >> 16) & 1u);           // round-to-nearest-even
  return (unsigned short)(u >> 16);
}

// ---------------- WMMA fragment loaders (bf16, 16x16x32) ----------------
// A matrix 16x32 (MxK), ISA 7.12.2: lanes0-15 K={0..7,16..23}, lanes16-31 K={8..15,24..31}
__device__ __forceinline__ v16bf load_frag_a(const unsigned short* __restrict__ base,
                                             int ld, int r0, int k0, int lane) {
  const int rr  = r0 + (lane & 15);
  const int kb2 = (lane >> 4) << 2;          // k-offset/2: 0 or 4
  const unsigned* p = (const unsigned*)(base + (long long)rr * ld + k0);
  union { unsigned u[8]; v16bf v; } f;
#pragma unroll
  for (int i = 0; i < 4; ++i) { f.u[i] = p[kb2 + i]; f.u[4 + i] = p[kb2 + 8 + i]; }
  return f.v;
}
// B matrix 32x16 (KxN) stored as [N,K]: lanes0-15 K={0..7,16..23}, lanes16-31 K={8..15,24..31}
__device__ __forceinline__ v16bf load_frag_b(const unsigned short* __restrict__ base,
                                             int ld, int n0, int k0, int lane) {
  const int nn  = n0 + (lane & 15);
  const int kb2 = (lane >> 4) << 2;          // k-offset/2: 0 or 4
  const unsigned* p = (const unsigned*)(base + (long long)nn * ld + k0);
  union { unsigned u[8]; v16bf v; } f;
#pragma unroll
  for (int i = 0; i < 4; ++i) { f.u[i] = p[kb2 + i]; f.u[4 + i] = p[kb2 + 8 + i]; }
  return f.v;
}

// ---------------- batched-strided bf16 WMMA GEMM ----------------
// C[m,n] = scale * sum_k A[m,k]*B^T[n,k] + bias[n]; optional relu; fp32 and/or bf16 out
// Block: 256 threads = 8 waves (4 in M x 2 in N); block tile 256x64; wave tile 64x32.
struct GemmArgs {
  const unsigned short* A;
  const unsigned short* B;   // stored [N,K]
  const float* bias;
  float* C;
  unsigned short* Dbf;
  long long sAb, sAh, sBb, sBh, sCb, sCh, sDb, sDh, sbh;
  int lda, ldb, ldc, ldd;
  int M, N, K, nH;
  float scale;
  int relu, transD;          // transD: write Dbf[n*ldd + m] (for V^T)
};

__global__ __launch_bounds__(256) void k_gemm(GemmArgs g) {
  const int lane = threadIdx.x & 31;
  const int w    = threadIdx.x >> 5;
  const int wm   = w & 3, wn = w >> 2;       // 4 waves in M, 2 in N
  const int z    = blockIdx.z;
  const int zb   = z / g.nH, zh = z % g.nH;
  const unsigned short* A = g.A + zb * g.sAb + zh * g.sAh;
  const unsigned short* B = g.B + zb * g.sBb + zh * g.sBh;
  const int m0 = blockIdx.x * 256 + wm * 64;
  const int n0 = blockIdx.y * 64  + wn * 32;

  v8f acc[4][2] = {};
  for (int k0 = 0; k0 < g.K; k0 += 32) {
    v16bf b0 = load_frag_b(B, g.ldb, n0,      k0, lane);
    v16bf b1 = load_frag_b(B, g.ldb, n0 + 16, k0, lane);
#pragma unroll
    for (int i = 0; i < 4; ++i) {
      v16bf a = load_frag_a(A, g.lda, m0 + i * 16, k0, lane);
      acc[i][0] = __builtin_amdgcn_wmma_f32_16x16x32_bf16(false, a, false, b0, (short)0, acc[i][0], false, false);
      acc[i][1] = __builtin_amdgcn_wmma_f32_16x16x32_bf16(false, a, false, b1, (short)0, acc[i][1], false, false);
    }
  }

  const float* bias = g.bias ? g.bias + zh * g.sbh : nullptr;
  float* C = g.C ? g.C + zb * g.sCb + zh * g.sCh : nullptr;
  unsigned short* Dp = g.Dbf ? g.Dbf + zb * g.sDb + zh * g.sDh : nullptr;
  const int nl = lane & 15;
  const int mb = (lane >> 4) << 3;
#pragma unroll
  for (int i = 0; i < 4; ++i) {
#pragma unroll
    for (int j = 0; j < 2; ++j) {
      v8f a = acc[i][j];
      const int gn = n0 + j * 16 + nl;
      const float bv = bias ? bias[gn] : 0.f;
#pragma unroll
      for (int v = 0; v < 8; ++v) {
        const int gm = m0 + i * 16 + mb + v;
        float val = a[v] * g.scale + bv;
        if (g.relu) val = fmaxf(val, 0.f);
        if (C) C[(long long)gm * g.ldc + gn] = val;
        if (Dp) {
          if (g.transD) Dp[(long long)gn * g.ldd + gm] = f2bf(val);
          else          Dp[(long long)gm * g.ldd + gn] = f2bf(val);
        }
      }
    }
  }
}

// ---------------- weight transpose+convert: fp32 [K,N] -> bf16 [N,K] ----------------
__global__ void k_transpose(const float* __restrict__ in, unsigned short* __restrict__ out,
                            int K, int N) {
  int idx = blockIdx.x * blockDim.x + threadIdx.x;
  if (idx >= K * N) return;
  int k = idx / N, n = idx % N;
  out[(long long)n * K + k] = f2bf(in[idx]);
}

// ---------------- embedding + positional encoding ----------------
__global__ void k_embed(const int* __restrict__ tok, const float* __restrict__ emb,
                        const float* __restrict__ pe, float* __restrict__ x,
                        unsigned short* __restrict__ xb) {
  const int r = blockIdx.x;                  // token row 0..MM-1
  const int t = tok[r];
  const int s = r % SSQ;
  for (int d = threadIdx.x; d < DD; d += blockDim.x) {
    float v = emb[(long long)t * DD + d] + pe[(long long)s * DD + d];
    x[(long long)r * DD + d]  = v;
    xb[(long long)r * DD + d] = f2bf(v);
  }
}

// ---------------- row softmax (optionally causal) -> bf16 ----------------
// One block per row; S=512, 256 threads -> 2 values per thread kept in registers
// (single global read pass instead of three).
__global__ __launch_bounds__(256) void k_softmax(const float* __restrict__ sc,
                                                 unsigned short* __restrict__ out, int causal) {
  const int r = blockIdx.x;                  // r = z*S + q
  const int q = r % SSQ;
  const int tid = threadIdx.x;
  const float* row = sc + (long long)r * SSQ;
  unsigned short* orow = out + (long long)r * SSQ;
  float v0 = row[tid];
  float v1 = row[tid + 256];
  if (causal) {
    if (tid > q)       v0 = -1e30f;
    if (tid + 256 > q) v1 = -1e30f;
  }
  __shared__ float red[256];
  red[tid] = fmaxf(v0, v1); __syncthreads();
  for (int o = 128; o; o >>= 1) {
    if (tid < o) red[tid] = fmaxf(red[tid], red[tid + o]);
    __syncthreads();
  }
  const float m = red[0]; __syncthreads();
  const float e0 = __expf(v0 - m), e1 = __expf(v1 - m);
  red[tid] = e0 + e1; __syncthreads();
  for (int o = 128; o; o >>= 1) {
    if (tid < o) red[tid] += red[tid + o];
    __syncthreads();
  }
  const float inv = 1.0f / red[0];
  orow[tid]       = f2bf(e0 * inv);
  orow[tid + 256] = f2bf(e1 * inv);
}

// ---------------- residual add + LayerNorm -> fp32 stream + bf16 copy ----------------
// One block per row; D=512, 256 threads -> 2 values/thread in registers (one read pass).
__global__ __launch_bounds__(256) void k_add_ln(float* __restrict__ x, const float* __restrict__ t,
                                                const float* __restrict__ gam,
                                                const float* __restrict__ bet,
                                                unsigned short* __restrict__ xb) {
  const int r = blockIdx.x;
  const int tid = threadIdx.x;
  float* xr = x + (long long)r * DD;
  const float* tr = t + (long long)r * DD;
  const float h0 = xr[tid] + tr[tid];
  const float h1 = xr[tid + 256] + tr[tid + 256];
  __shared__ float red[256];
  red[tid] = h0 + h1; __syncthreads();
  for (int o = 128; o; o >>= 1) { if (tid < o) red[tid] += red[tid + o]; __syncthreads(); }
  const float mu = red[0] / (float)DD; __syncthreads();
  red[tid] = h0 * h0 + h1 * h1; __syncthreads();
  for (int o = 128; o; o >>= 1) { if (tid < o) red[tid] += red[tid + o]; __syncthreads(); }
  const float var = red[0] / (float)DD - mu * mu;
  const float rstd = rsqrtf(var + 1e-5f);
  const float y0 = gam[tid] * (h0 - mu) * rstd + bet[tid];
  const float y1 = gam[tid + 256] * (h1 - mu) * rstd + bet[tid + 256];
  xr[tid] = y0;
  xr[tid + 256] = y1;
  xb[(long long)r * DD + tid]       = f2bf(y0);
  xb[(long long)r * DD + tid + 256] = f2bf(y1);
}

// ---------------- input index map (setup_inputs dict, insertion order) ----------------
enum {
  I_SRC = 0, I_TGT, I_SRCMASK, I_TGTMASK,
  I_ENC_EMB, I_DEC_EMB,
  E_WQ, E_BQ, E_WK, E_BK, E_WV, E_BV, E_WO, E_BO,
  E_FF1W, E_FF1B, E_FF2W, E_FF2B,
  E_LN1G, E_LN1B, E_LN2G, E_LN2B,
  Dc_SWQ, Dc_SBQ, Dc_SWK, Dc_SBK, Dc_SWV, Dc_SBV, Dc_SWO, Dc_SBO,
  Dc_CWQ, Dc_CBQ, Dc_CWK, Dc_CBK, Dc_CWV, Dc_CBV, Dc_CWO, Dc_CBO,
  Dc_FF1W, Dc_FF1B, Dc_FF2W, Dc_FF2B,
  Dc_LN1G, Dc_LN1B, Dc_LN2G, Dc_LN2B, Dc_LN3G, Dc_LN3B,
  I_OUTW, I_OUTB, I_PE
};

extern "C" void kernel_launch(void* const* d_in, const int* in_sizes, int n_in,
                              void* d_out, int out_size, void* d_ws, size_t ws_size,
                              hipStream_t stream) {
  (void)in_sizes; (void)n_in; (void)out_size; (void)ws_size;
  const int*   src     = (const int*)d_in[I_SRC];
  const int*   tgt     = (const int*)d_in[I_TGT];
  const float* enc_emb = (const float*)d_in[I_ENC_EMB];
  const float* dec_emb = (const float*)d_in[I_DEC_EMB];
  const float* pe      = (const float*)d_in[I_PE];
  const float* out_w   = (const float*)d_in[I_OUTW];
  const float* out_b   = (const float*)d_in[I_OUTB];

  // -------- workspace layout (bump allocator, ~162 MB) --------
  char* p = (char*)d_ws;
  auto alloc = [&](size_t n) { char* r = p; p += (n + 255) & ~(size_t)255; return r; };
  float*          x      = (float*)         alloc((size_t)MM * DD * 4);
  unsigned short* xb     = (unsigned short*)alloc((size_t)MM * DD * 2);
  unsigned short* membf  = (unsigned short*)alloc((size_t)MM * DD * 2);
  unsigned short* qbf    = (unsigned short*)alloc((size_t)MM * DD * 2);
  unsigned short* kbf    = (unsigned short*)alloc((size_t)MM * DD * 2);
  unsigned short* vT     = (unsigned short*)alloc((size_t)BBATCH * HH * DH * SSQ * 2);
  float*          scores = (float*)         alloc((size_t)BBATCH * HH * SSQ * SSQ * 4);
  unsigned short* attnbf = (unsigned short*)alloc((size_t)BBATCH * HH * SSQ * SSQ * 2);
  unsigned short* ctxbf  = (unsigned short*)alloc((size_t)MM * DD * 2);
  float*          tbuf   = (float*)         alloc((size_t)MM * DD * 4);
  unsigned short* h1bf   = (unsigned short*)alloc((size_t)MM * FF * 2);
  unsigned short* wT     = (unsigned short*)alloc((size_t)FF * DD * 2);  // per-GEMM weight scratch

  auto T = [&](const float* w, int K, int N) {
    int n = K * N;
    k_transpose<<<dim3((n + 255) / 256), dim3(256), 0, stream>>>(w, wT, K, N);
  };
  auto rungemm = [&](const GemmArgs& g, int nb) {
    dim3 grid(g.M / 256, g.N / 64, nb);
    k_gemm<<<grid, dim3(256), 0, stream>>>(g);
  };
  // plain projection: [MM,K] @ W[K,N] + b  (A row stride = K)
  auto proj = [&](const unsigned short* A, const float* w, const float* b, int K, int N,
                  float* C, unsigned short* Dbf, int relu) {
    T(w, K, N);
    GemmArgs g{};
    g.A = A; g.lda = K; g.B = wT; g.ldb = K; g.bias = b;
    g.C = C; g.ldc = N; g.Dbf = Dbf; g.ldd = N;
    g.M = MM; g.N = N; g.K = K; g.nH = 1; g.scale = 1.f; g.relu = relu;
    rungemm(g, 1);
  };

  auto do_attn = [&](const unsigned short* qsrc, const unsigned short* kvsrc,
                     const float* wq, const float* bq, const float* wk, const float* bk,
                     const float* wv, const float* bv, const float* wo, const float* bo,
                     int causal) {
    proj(qsrc,  wq, bq, DD, DD, nullptr, qbf, 0);
    proj(kvsrc, wk, bk, DD, DD, nullptr, kbf, 0);
    // V projection batched over (b,h), epilogue writes V^T [B,H,dh,S]
    T(wv, DD, DD);
    {
      GemmArgs g{};
      g.A = kvsrc; g.sAb = (long long)SSQ * DD; g.lda = DD;
      g.B = wT;    g.sBh = (long long)DH * DD;  g.ldb = DD;
      g.bias = bv; g.sbh = DH;
      g.Dbf = vT;  g.sDb = (long long)HH * DH * SSQ; g.sDh = (long long)DH * SSQ;
      g.ldd = SSQ; g.transD = 1;
      g.M = SSQ; g.N = DH; g.K = DD; g.nH = HH; g.scale = 1.f;
      rungemm(g, BBATCH * HH);
    }
    // scores[b,h] = Q K^T / sqrt(dh)
    {
      GemmArgs g{};
      g.A = qbf; g.sAb = (long long)SSQ * DD; g.sAh = DH; g.lda = DD;
      g.B = kbf; g.sBb = (long long)SSQ * DD; g.sBh = DH; g.ldb = DD;
      g.C = scores; g.sCb = (long long)HH * SSQ * SSQ; g.sCh = (long long)SSQ * SSQ; g.ldc = SSQ;
      g.M = SSQ; g.N = SSQ; g.K = DH; g.nH = HH; g.scale = 0.125f;
      rungemm(g, BBATCH * HH);
    }
    k_softmax<<<dim3(BBATCH * HH * SSQ), dim3(256), 0, stream>>>(scores, attnbf, causal);
    // ctx[b,h] = attn @ V  -> packed back to [B*S, D]
    {
      GemmArgs g{};
      g.A = attnbf; g.sAb = (long long)HH * SSQ * SSQ; g.sAh = (long long)SSQ * SSQ; g.lda = SSQ;
      g.B = vT;     g.sBb = (long long)HH * DH * SSQ;  g.sBh = (long long)DH * SSQ;  g.ldb = SSQ;
      g.Dbf = ctxbf; g.sDb = (long long)SSQ * DD; g.sDh = DH; g.ldd = DD;
      g.M = SSQ; g.N = DH; g.K = SSQ; g.nH = HH; g.scale = 1.f;
      rungemm(g, BBATCH * HH);
    }
    proj(ctxbf, wo, bo, DD, DD, tbuf, nullptr, 0);
  };

  auto lnorm = [&](const float* gam, const float* bet) {
    k_add_ln<<<dim3(MM), dim3(256), 0, stream>>>(x, tbuf, gam, bet, xb);
  };
  auto do_ffn = [&](const float* w1, const float* b1, const float* w2, const float* b2) {
    proj(xb,   w1, b1, DD, FF, nullptr, h1bf, 1);
    proj(h1bf, w2, b2, FF, DD, tbuf, nullptr, 0);
  };
  auto W = [&](int idx, size_t per) { return (const float*)d_in[idx] + per; };

  // ================= encoder =================
  k_embed<<<dim3(MM), dim3(256), 0, stream>>>(src, enc_emb, pe, x, xb);
  for (int l = 0; l < LL; ++l) {
    const size_t wdd = (size_t)l * DD * DD, wdf = (size_t)l * DD * FF;
    do_attn(xb, xb,
            W(E_WQ, wdd), W(E_BQ, (size_t)l * DD), W(E_WK, wdd), W(E_BK, (size_t)l * DD),
            W(E_WV, wdd), W(E_BV, (size_t)l * DD), W(E_WO, wdd), W(E_BO, (size_t)l * DD), 0);
    lnorm(W(E_LN1G, (size_t)l * DD), W(E_LN1B, (size_t)l * DD));
    do_ffn(W(E_FF1W, wdf), W(E_FF1B, (size_t)l * FF), W(E_FF2W, wdf), W(E_FF2B, (size_t)l * DD));
    lnorm(W(E_LN2G, (size_t)l * DD), W(E_LN2B, (size_t)l * DD));
  }
  hipMemcpyAsync(membf, xb, (size_t)MM * DD * 2, hipMemcpyDeviceToDevice, stream);

  // ================= decoder =================
  k_embed<<<dim3(MM), dim3(256), 0, stream>>>(tgt, dec_emb, pe, x, xb);
  for (int l = 0; l < LL; ++l) {
    const size_t wdd = (size_t)l * DD * DD, wdf = (size_t)l * DD * FF, wd = (size_t)l * DD;
    do_attn(xb, xb,
            W(Dc_SWQ, wdd), W(Dc_SBQ, wd), W(Dc_SWK, wdd), W(Dc_SBK, wd),
            W(Dc_SWV, wdd), W(Dc_SBV, wd), W(Dc_SWO, wdd), W(Dc_SBO, wd), 1);
    lnorm(W(Dc_LN1G, wd), W(Dc_LN1B, wd));
    do_attn(xb, membf,
            W(Dc_CWQ, wdd), W(Dc_CBQ, wd), W(Dc_CWK, wdd), W(Dc_CBK, wd),
            W(Dc_CWV, wdd), W(Dc_CBV, wd), W(Dc_CWO, wdd), W(Dc_CBO, wd), 0);
    lnorm(W(Dc_LN2G, wd), W(Dc_LN2B, wd));
    do_ffn(W(Dc_FF1W, wdf), W(Dc_FF1B, (size_t)l * FF), W(Dc_FF2W, wdf), W(Dc_FF2B, wd));
    lnorm(W(Dc_LN3G, wd), W(Dc_LN3B, wd));
  }

  // ================= output projection -> fp32 logits =================
  T(out_w, DD, VV);
  {
    GemmArgs g{};
    g.A = xb; g.lda = DD; g.B = wT; g.ldb = DD; g.bias = out_b;
    g.C = (float*)d_out; g.ldc = VV;
    g.M = MM; g.N = VV; g.K = DD; g.nH = 1; g.scale = 1.f;
    rungemm(g, 1);
  }
}
